// APPNP_model_74277164417193
// MI455X (gfx1250) — matrix operand order repaired and checked by reading, not compiled, not run on hardware
//
#include <hip/hip_runtime.h>
#include <hip/hip_bf16.h>

typedef __attribute__((ext_vector_type(16))) _Float16 v16h;
typedef __attribute__((ext_vector_type(8)))  _Float16 v8h;
typedef __attribute__((ext_vector_type(8)))  float    v8f;

#define N_NODES 100000
#define N_EDGES 3200000
#define K_IN    1433
#define K_PAD   1440        // 45 * 32, zero-padded
#define HID     64
#define NCLS    7
#define ALPHA   0.1f
#define K_STEPS 10

// ---------------------------------------------------------------------------
// Pack W1 [1433,64] f32 -> W1T [64][1440] f16, transposed + zero-padded.
// One-time, tiny (184 KB). Lets the GEMM stage B tiles with aligned b128 ops.
// ---------------------------------------------------------------------------
__global__ __launch_bounds__(256) void pack_w1(const float* __restrict__ W1,
                                               _Float16* __restrict__ W1T) {
    int idx = blockIdx.x * 256 + threadIdx.x;       // over 64*1440
    if (idx >= HID * K_PAD) return;
    int n = idx / K_PAD, k = idx % K_PAD;
    float v = (k < K_IN) ? W1[(size_t)k * HID + n] : 0.0f;
    W1T[idx] = (_Float16)v;
}

// ---------------------------------------------------------------------------
// GEMM1: H1 = relu(x @ W1 + b1)
// 256 threads = 8 waves; block computes 128 rows x 64 cols.
// Each wave: 16 rows x 64 cols via 4x v_wmma_f32_16x16x32_f16.
// LDS B tile double-buffered: 1 barrier per K-block; guarded tail peeled.
// ---------------------------------------------------------------------------
__global__ __launch_bounds__(256) void gemm1_wmma(
    const float* __restrict__ x, const _Float16* __restrict__ W1T,
    const float* __restrict__ b1, float* __restrict__ H1)
{
    __shared__ __align__(16) _Float16 sB[2][64][32];   // [buf][n][k] 8 KB

    const int lane    = threadIdx.x & 31;
    const int wave    = threadIdx.x >> 5;
    const int c16     = lane & 15;
    const int khalf   = lane >> 4;         // 0: lanes 0-15, 1: lanes 16-31
    const int rowbase = blockIdx.x * 128 + wave * 16;

    int arow = rowbase + c16;              // A row; clamp (EXEC stays all-ones)
    if (arow >= N_NODES) arow = N_NODES - 1;
    const float* xrow = x + (size_t)arow * K_IN;

    // staging: thread t copies 8 f16 (16 B) of W1T column sn, K chunk sk
    const int sn = threadIdx.x >> 2;           // 0..63
    const int sk = (threadIdx.x & 3) * 8;      // 0,8,16,24
    const _Float16* wsrc = W1T + (size_t)sn * K_PAD + sk;

    v8f acc[4];
#pragma unroll
    for (int nt = 0; nt < 4; ++nt) acc[nt] = (v8f){0,0,0,0,0,0,0,0};

    // prologue: stage block 0
    *(v8h*)&sB[0][sn][sk] = *(const v8h*)(wsrc);
    __syncthreads();

    const int KBLK = K_PAD / 32;               // 45 (last block guarded)
    for (int kb = 0; kb < KBLK; ++kb) {
        const int cur = kb & 1;
        const int k0  = kb * 32;

        // stage next B tile into the idle buffer (padded W1T: no guard)
        if (kb < KBLK - 1)
            *(v8h*)&sB[cur ^ 1][sn][sk] = *(const v8h*)(wsrc + (kb + 1) * 32);

        // A fragment: 16x32 f16, ISA layout
        // lanes 0-15: K 0-7 (V0-3), 16-23 (V4-7); lanes 16-31: 8-15, 24-31
        v16h a;
        if (kb < KBLK - 1) {                    // unguarded fast path
            const float* p0 = xrow + k0 + khalf * 8;
#pragma unroll
            for (int j = 0; j < 8; ++j) {
                a[j]     = (_Float16)p0[j];
                a[8 + j] = (_Float16)p0[16 + j];
            }
            __builtin_prefetch(p0 + 32, 0, 3);  // next K-block of this row
        } else {                                // peeled tail: zero-fill K>=1433
#pragma unroll
            for (int j = 0; j < 8; ++j) {
                int k1 = k0 + khalf * 8 + j;
                int k2 = k0 + 16 + khalf * 8 + j;
                a[j]     = (_Float16)((k1 < K_IN) ? xrow[k1] : 0.0f);
                a[8 + j] = (_Float16)((k2 < K_IN) ? xrow[k2] : 0.0f);
            }
        }

#pragma unroll
        for (int nt = 0; nt < 4; ++nt) {
            // B lane c (0-15): K 0-7, 16-23; lanes 16-31: K 8-15, 24-31
            v8h b0 = *(const v8h*)&sB[cur][nt * 16 + c16][khalf * 8];
            v8h b1v = *(const v8h*)&sB[cur][nt * 16 + c16][16 + khalf * 8];
            v16h b = __builtin_shufflevector(b0, b1v, 0,1,2,3,4,5,6,7,8,9,10,11,12,13,14,15);
            acc[nt] = __builtin_amdgcn_wmma_f32_16x16x32_f16(
                false, a, false, b, (short)0, acc[nt], false, false);
        }
        __syncthreads();   // publishes next buffer + protects current one
    }

    // epilogue: + b1, relu, store. C/D: lanes<16 -> M 0-7, lanes>=16 -> M 8-15
    const int rowoff = khalf * 8;
#pragma unroll
    for (int nt = 0; nt < 4; ++nt) {
        int col = nt * 16 + c16;
        float bias = b1[col];
#pragma unroll
        for (int r = 0; r < 8; ++r) {
            int row = rowbase + rowoff + r;
            if (row < N_NODES) {
                float v = acc[nt][r] + bias;
                H1[(size_t)row * HID + col] = v > 0.0f ? v : 0.0f;
            }
        }
    }
}

// ---------------------------------------------------------------------------
// GEMM2: h = H1 @ W2 + b2  -> writes h0 and cur (stride-8 padded rows)
// ---------------------------------------------------------------------------
__global__ __launch_bounds__(256) void gemm2(
    const float* __restrict__ H1, const float* __restrict__ W2,
    const float* __restrict__ b2, float* __restrict__ h0, float* __restrict__ cur)
{
    __shared__ float sW[HID * NCLS];
    __shared__ float sb[NCLS];
    for (int i = threadIdx.x; i < HID * NCLS; i += 256) sW[i] = W2[i];
    if (threadIdx.x < NCLS) sb[threadIdx.x] = b2[threadIdx.x];
    __syncthreads();

    int node = blockIdx.x * 256 + threadIdx.x;
    if (node >= N_NODES) return;

    float acc[NCLS];
#pragma unroll
    for (int c = 0; c < NCLS; ++c) acc[c] = sb[c];
    const float* hrow = H1 + (size_t)node * HID;
#pragma unroll 4
    for (int k = 0; k < HID; ++k) {
        float v = hrow[k];
#pragma unroll
        for (int c = 0; c < NCLS; ++c) acc[c] = fmaf(v, sW[k * NCLS + c], acc[c]);
    }
    float* o1 = h0  + (size_t)node * 8;
    float* o2 = cur + (size_t)node * 8;
#pragma unroll
    for (int c = 0; c < NCLS; ++c) { o1[c] = acc[c]; o2[c] = acc[c]; }
    o1[7] = 0.0f; o2[7] = 0.0f;
}

// ---------------------------------------------------------------------------
// Degree / normalization
// ---------------------------------------------------------------------------
__global__ void deg_init(float* deg) {
    int i = blockIdx.x * 256 + threadIdx.x;
    if (i < N_NODES) deg[i] = 1.0f;               // self-loop
}
__global__ void deg_count(const int* __restrict__ col, float* deg) {
    int e = blockIdx.x * 256 + threadIdx.x;
    if (e < N_EDGES) atomicAdd(&deg[col[e]], 1.0f);
}
__global__ void deg_rsqrt(float* deg) {
    int i = blockIdx.x * 256 + threadIdx.x;
    if (i < N_NODES) deg[i] = rsqrtf(deg[i]);     // deg >= 1 always
}
__global__ void edge_weight(const int* __restrict__ row, const int* __restrict__ col,
                            const float* __restrict__ dinv, float* __restrict__ wedge) {
    int e = blockIdx.x * 256 + threadIdx.x;
    if (e < N_EDGES) wedge[e] = dinv[row[e]] * dinv[col[e]];
}

// ---------------------------------------------------------------------------
// Propagation step: nxt = alpha*h0 + (1-alpha)*(self + edge aggregation)
// ---------------------------------------------------------------------------
__global__ void step_init(const float* __restrict__ cur, const float* __restrict__ h0,
                          const float* __restrict__ dinv, float* __restrict__ nxt) {
    int i = blockIdx.x * 256 + threadIdx.x;
    if (i >= N_NODES) return;
    float w = (1.0f - ALPHA) * dinv[i] * dinv[i];     // self-loop norm
    const float* c = cur + (size_t)i * 8;
    const float* h = h0  + (size_t)i * 8;
    float*       n = nxt + (size_t)i * 8;
#pragma unroll
    for (int ch = 0; ch < NCLS; ++ch) n[ch] = ALPHA * h[ch] + w * c[ch];
    n[7] = 0.0f;
}
__global__ void step_scatter(const int* __restrict__ row, const int* __restrict__ col,
                             const float* __restrict__ wedge,
                             const float* __restrict__ cur, float* __restrict__ nxt) {
    int e = blockIdx.x * 256 + threadIdx.x;
    if (e >= N_EDGES) return;
    int r = row[e], c = col[e];
    float w = (1.0f - ALPHA) * wedge[e];
    const float* src = cur + (size_t)r * 8;           // 3.2 MB -> L2 resident
    float*       dst = nxt + (size_t)c * 8;
#pragma unroll
    for (int ch = 0; ch < NCLS; ++ch) atomicAdd(&dst[ch], w * src[ch]);
}

// ---------------------------------------------------------------------------
// log_softmax over 7 classes -> d_out [N,7]
// ---------------------------------------------------------------------------
__global__ void logsm(const float* __restrict__ cur, float* __restrict__ out) {
    int i = blockIdx.x * 256 + threadIdx.x;
    if (i >= N_NODES) return;
    const float* v = cur + (size_t)i * 8;
    float m = v[0];
#pragma unroll
    for (int c = 1; c < NCLS; ++c) m = fmaxf(m, v[c]);
    float s = 0.0f;
#pragma unroll
    for (int c = 0; c < NCLS; ++c) s += __expf(v[c] - m);
    float lse = __logf(s);
#pragma unroll
    for (int c = 0; c < NCLS; ++c) out[(size_t)i * NCLS + c] = v[c] - m - lse;
}

// ---------------------------------------------------------------------------
extern "C" void kernel_launch(void* const* d_in, const int* in_sizes, int n_in,
                              void* d_out, int out_size, void* d_ws, size_t ws_size,
                              hipStream_t stream) {
    const float* x   = (const float*)d_in[0];
    const int*   ei  = (const int*)  d_in[1];   // [2, E]: row=src, col=dst
    const float* W1  = (const float*)d_in[2];
    const float* b1  = (const float*)d_in[3];
    const float* W2  = (const float*)d_in[4];
    const float* b2  = (const float*)d_in[5];
    float* out = (float*)d_out;

    const int* erow = ei;
    const int* ecol = ei + N_EDGES;

    // workspace layout (in floats)
    float*    ws    = (float*)d_ws;
    _Float16* W1T   = (_Float16*)ws;                       // 64*1440 f16 = 46080 f32
    float*    H1    = ws + 46336;                          // 100000*64
    float*    h0    = H1  + (size_t)N_NODES * HID;         // 800,000
    float*    bufA  = h0  + (size_t)N_NODES * 8;
    float*    bufB  = bufA + (size_t)N_NODES * 8;
    float*    dinv  = bufB + (size_t)N_NODES * 8;          // 100,096 (padded)
    float*    wedge = dinv + 100096;                       // 3,200,000

    const int NB_NODE = (N_NODES + 255) / 256;             // 391
    const int NB_EDGE = (N_EDGES + 255) / 256;             // 12500
    const int NB_GEMM = (N_NODES + 127) / 128;             // 782
    const int NB_PACK = (HID * K_PAD + 255) / 256;         // 360

    // 0) pack W1 -> f16 transposed, padded
    pack_w1<<<NB_PACK, 256, 0, stream>>>(W1, W1T);

    // 1) H1 = relu(x @ W1 + b1)   (WMMA)
    gemm1_wmma<<<NB_GEMM, 256, 0, stream>>>(x, W1T, b1, H1);

    // 2) h0 = cur = H1 @ W2 + b2
    gemm2<<<NB_NODE, 256, 0, stream>>>(H1, W2, b2, h0, bufA);

    // 3) normalization: deg -> dinv -> per-edge weights
    deg_init <<<NB_NODE, 256, 0, stream>>>(dinv);
    deg_count<<<NB_EDGE, 256, 0, stream>>>(ecol, dinv);
    deg_rsqrt<<<NB_NODE, 256, 0, stream>>>(dinv);
    edge_weight<<<NB_EDGE, 256, 0, stream>>>(erow, ecol, dinv, wedge);

    // 4) K_STEPS of propagation, double-buffered (fixed sequence: capture-safe)
    float* cur = bufA;
    float* nxt = bufB;
    for (int s = 0; s < K_STEPS; ++s) {
        step_init   <<<NB_NODE, 256, 0, stream>>>(cur, h0, dinv, nxt);
        step_scatter<<<NB_EDGE, 256, 0, stream>>>(erow, ecol, wedge, cur, nxt);
        float* t = cur; cur = nxt; nxt = t;
    }

    // 5) log_softmax -> out
    logsm<<<NB_NODE, 256, 0, stream>>>(cur, out);
}
